// GRUD_53772990546344
// MI455X (gfx1250) — matrix operand order repaired and checked by reading, not compiled, not run on hardware
//
#include <hip/hip_runtime.h>

// ---------------------------------------------------------------------------
// GRU-D (B=32, F=128, H=256, T=2048) for MI455X / gfx1250.
// The batch recurrences are independent, so the problem splits into TWO
// communication-free persistent workgroups, each owning 16 batch rows
// (one WMMA M-tile) with 16 wave32 waves (one per 16-column tile of H).
// All matmuls via v_wmma_f32_16x16x32_bf16; weights pre-packed to bf16 tiles
// streamed from L2 (LICM-defeated so they are not hoisted+spilled to scratch).
// ---------------------------------------------------------------------------

#define B_ 32
#define F_ 128
#define H_ 256
#define T_ 2048
#define MB_ 16            // batch rows per block (one WMMA M-tile)

typedef __attribute__((ext_vector_type(16))) __bf16 v16bf;
typedef __attribute__((ext_vector_type(8)))  float  v8f;

union ABf { v16bf v; unsigned u[8]; };

__device__ __forceinline__ unsigned short f2bf(float f) {
  unsigned u = __float_as_uint(f);
  unsigned r = u + 0x7FFFu + ((u >> 16) & 1u);   // round-to-nearest-even
  return (unsigned short)(r >> 16);
}
__device__ __forceinline__ float bf2f(unsigned short h) {
  return __uint_as_float(((unsigned)h) << 16);
}

// ---- packed-weight workspace layout (dword offsets) -----------------------
// tile = 32(K) x 16(N) bf16, 256 dwords. Within a tile, dword index =
// lane*8 + v  (each lane reads 32 contiguous bytes -> 2x global_load_b128),
// holding B[k][n], B[k+1][n] with k = kt*32 + (v<4 ? 2v : 16+2(v-4)) + (lane>=16)*8,
// n = nt*16 + (lane&15).
#define OFF_WDH0 0u        // K=128 N=256 ->  64 tiles
#define OFF_W0   16384u    // K=128 N=768 -> 192 tiles
#define OFF_V0   65536u    // K=128 N=768
#define OFF_U0   114688u   // K=256 N=768 -> 384 tiles
#define OFF_WDH1 212992u   // K=128 N=256
#define OFF_W1   229376u   // K=256 N=768
#define OFF_V1   327680u   // K=128 N=768
#define OFF_U1   376832u   // K=256 N=768

__global__ __launch_bounds__(256)
void pack_bf16_tiles(const float* __restrict__ src, int K, int N, unsigned* __restrict__ dst) {
  int nT   = N >> 4;
  int tile = blockIdx.x;
  int kt = tile / nT, nt = tile % nT;
  int tid = threadIdx.x;               // = v*32 + lane
  int v = tid >> 5, lane = tid & 31;
  int k = kt * 32 + ((v < 4) ? 2 * v : 16 + 2 * (v - 4)) + ((lane >> 4) << 3);
  int n = nt * 16 + (lane & 15);
  unsigned lo = f2bf(src[(size_t)k * N + n]);
  unsigned hi = f2bf(src[(size_t)(k + 1) * N + n]);
  dst[(size_t)tile * 256u + (size_t)lane * 8u + v] = lo | (hi << 16);
}

// ---- WMMA helpers ---------------------------------------------------------
// A 16x32 bf16 layout: m=lane&15, k = (v<4 ? 2v : 16+2(v-4)) + (lane>=16)*8
template <int K>
__device__ __forceinline__ v16bf load_a(const unsigned short* __restrict__ lds,
                                        int ktile, int lane) {
  ABf r;
  int m = lane & 15;
  int khalf = (lane >> 4) << 3;
  const unsigned short* row = lds + m * K + ktile * 32;
#pragma unroll
  for (int v = 0; v < 8; ++v) {
    int k = (v < 4) ? (2 * v + khalf) : (16 + 2 * (v - 4) + khalf);
    r.u[v] = *(const unsigned*)(row + k);   // dword-aligned: K even, k even
  }
  return r.v;
}

// B tile: 2x global_load_b128 per lane from L2-resident packed weights.
// The empty asm makes the offset opaque so LLVM cannot hoist these
// loop-invariant loads out of the 2048-step loop (scratch-spill hazard).
__device__ __forceinline__ v16bf load_b(const unsigned* __restrict__ mat,
                                        int tileIdx, int lane) {
  int off = tileIdx * 256 + lane * 8;
  asm("" : "+v"(off));                 // opaque: stream from L2 every step
  ABf r;
  const unsigned* p = mat + off;
#pragma unroll
  for (int v = 0; v < 8; ++v) r.u[v] = p[v];
  return r.v;
}

#define WMMA_BF16(a, b, c) \
  __builtin_amdgcn_wmma_f32_16x16x32_bf16(false, (a), false, (b), (short)0, (c), false, false)

// single-output-tile GEMM, B double-buffered across K
template <int K>
__device__ __forceinline__ void gemm1(v8f& acc, const unsigned short* __restrict__ aLds,
                                      const unsigned* __restrict__ bMat, int nTiles,
                                      int nt, int lane) {
  v16bf b = load_b(bMat, 0 * nTiles + nt, lane);
#pragma unroll
  for (int kt = 0; kt < K / 32; ++kt) {
    v16bf cb = b;
    if (kt + 1 < K / 32) b = load_b(bMat, (kt + 1) * nTiles + nt, lane);
    v16bf a = load_a<K>(aLds, kt, lane);
    acc = WMMA_BF16(a, cb, acc);
  }
}

// two-output-tile GEMM sharing A tiles (r and z gates), B double-buffered
template <int K>
__device__ __forceinline__ void gemm2(v8f& acc0, v8f& acc1,
                                      const unsigned short* __restrict__ aLds,
                                      const unsigned* __restrict__ bMat, int nTiles,
                                      int nt0, int nt1, int lane) {
  v16bf b0 = load_b(bMat, 0 * nTiles + nt0, lane);
  v16bf b1 = load_b(bMat, 0 * nTiles + nt1, lane);
#pragma unroll
  for (int kt = 0; kt < K / 32; ++kt) {
    v16bf cb0 = b0, cb1 = b1;
    if (kt + 1 < K / 32) {
      b0 = load_b(bMat, (kt + 1) * nTiles + nt0, lane);
      b1 = load_b(bMat, (kt + 1) * nTiles + nt1, lane);
    }
    v16bf a = load_a<K>(aLds, kt, lane);
    acc0 = WMMA_BF16(a, cb0, acc0);
    acc1 = WMMA_BF16(a, cb1, acc1);
  }
}

// ---- one GRU-D cell step (16 waves cooperate; wave jt owns H-columns
// jt*16..jt*16+15). C/D f32 layout: acc[i] is element
// (m = 8*(lane>=16)+i, n = lane&15) of the 16x16 tile.
template <int KX>
__device__ __forceinline__ void grud_layer(
    const unsigned short* __restrict__ xin,  // (16, KX) bf16 A staging
    const unsigned short* __restrict__ mu,   // (16, 128)
    const unsigned short* __restrict__ du,   // (16, 128)
    unsigned short* __restrict__ h,          // (16, 256) state; overwritten with h_new
    unsigned short* __restrict__ hd,         // (16, 256) scratch: decayed h (bf16)
    unsigned short* __restrict__ rh,         // (16, 256) scratch: r * h_d (bf16)
    const unsigned* __restrict__ Wdh_p,
    const unsigned* __restrict__ W_p, const unsigned* __restrict__ V_p,
    const unsigned* __restrict__ U_p,
    float bdh, float br, float bz, float bt,  // per-wave bias scalars (preloaded)
    float* __restrict__ out_opt,              // optional (MB_,H) f32 output slice
    int jt, int lane) {
  const int n    = lane & 15;
  const int vrow = (lane >> 4) << 3;
  const int col  = jt * 16 + n;

  // (1) gamma_h = exp(-relu(delta @ Wdh + b_dh));  h_d = gamma_h * h
  v8f g = {};
  gemm1<128>(g, du, Wdh_p, H_ / 16, jt, lane);
  float hv[8];
#pragma unroll
  for (int i = 0; i < 8; ++i) hv[i] = bf2f(h[(vrow + i) * H_ + col]);  // clustered DS loads
  float hdv[8];
#pragma unroll
  for (int i = 0; i < 8; ++i) {
    float gam = __expf(-fmaxf(g[i] + bdh, 0.0f));
    hdv[i] = gam * hv[i];
    hd[(vrow + i) * H_ + col] = f2bf(hdv[i]);
  }
  __syncthreads();

  // (2) r, z gates (share A tiles):  pre = xin@W + m@V + h_d@U + b
  v8f r = {}, z = {};
  gemm2<KX>(r, z, xin, W_p, 48, jt, 16 + jt, lane);
  gemm2<128>(r, z, mu, V_p, 48, jt, 16 + jt, lane);
  gemm2<256>(r, z, hd, U_p, 48, jt, 16 + jt, lane);
  float zz[8];
#pragma unroll
  for (int i = 0; i < 8; ++i) {
    float rv = 1.0f / (1.0f + __expf(-(r[i] + br)));
    zz[i]    = 1.0f / (1.0f + __expf(-(z[i] + bz)));
    rh[(vrow + i) * H_ + col] = f2bf(rv * hdv[i]);
  }
  __syncthreads();

  // (3) h_tilde = tanh(xin@W3 + m@V3 + (r*h_d)@U3 + b3);  h_new blend
  v8f t = {};
  gemm1<KX>(t, xin, W_p, 48, 32 + jt, lane);
  gemm1<128>(t, mu, V_p, 48, 32 + jt, lane);
  gemm1<256>(t, rh, U_p, 48, 32 + jt, lane);
#pragma unroll
  for (int i = 0; i < 8; ++i) {
    float ht = tanhf(t[i] + bt);
    float hn = (1.0f - zz[i]) * hdv[i] + zz[i] * ht;
    h[(vrow + i) * H_ + col] = f2bf(hn);
    if (out_opt) out_opt[(vrow + i) * H_ + col] = hn;
  }
  __syncthreads();
}

// ---- persistent sequential kernel: 2 blocks x 512 threads (16 waves) ------
// Block bo owns batch rows [bo*16, bo*16+16): recurrences are independent.
__global__ __launch_bounds__(512)
void grud_seq(const float* __restrict__ input, const float* __restrict__ x_mean,
              const float* __restrict__ w_dx, const float* __restrict__ b_dx,
              const float* __restrict__ b_dh0, const float* __restrict__ b0,
              const float* __restrict__ b_dh1, const float* __restrict__ b1,
              const unsigned* __restrict__ ws, float* __restrict__ out) {
  __shared__ float          xl[MB_ * F_];        // 8 KB   x_last state
  __shared__ unsigned short h0u[MB_ * H_];       // 8 KB   bf16 h0 state
  __shared__ unsigned short h1u[MB_ * H_];       // 8 KB   bf16 h1 state
  __shared__ unsigned short hdu[MB_ * H_];       // 8 KB   decayed h scratch
  __shared__ unsigned short rhu[MB_ * H_];       // 8 KB   r*h_d scratch
  __shared__ unsigned short xhu[MB_ * F_];       // 4 KB   x_hat bf16
  __shared__ unsigned short mu[MB_ * F_];        // 4 KB   mask bf16
  __shared__ unsigned short du[MB_ * F_];        // 4 KB   delta bf16
  __shared__ float          xmS[F_], wdS[F_], bdS[F_];   // 1.5 KB params

  const int tid = threadIdx.x;
  const int lane = tid & 31, wave = tid >> 5;    // wave = jt in 0..15
  const int bo = blockIdx.x;                     // 0 or 1

  const unsigned* Wdh0p = ws + OFF_WDH0;
  const unsigned* W0p   = ws + OFF_W0;
  const unsigned* V0p   = ws + OFF_V0;
  const unsigned* U0p   = ws + OFF_U0;
  const unsigned* Wdh1p = ws + OFF_WDH1;
  const unsigned* W1p   = ws + OFF_W1;
  const unsigned* V1p   = ws + OFF_V1;
  const unsigned* U1p   = ws + OFF_U1;

  // per-wave bias scalars (loop-invariant; live in VGPRs for all 2048 steps)
  const int jn = wave * 16 + (lane & 15);
  const float bdh0v = b_dh0[jn];
  const float b0r = b0[jn], b0z = b0[H_ + jn], b0t = b0[2 * H_ + jn];
  const float bdh1v = b_dh1[jn];
  const float b1r = b1[jn], b1z = b1[H_ + jn], b1t = b1[2 * H_ + jn];

  // init: x_last = x_mean, h0 = h1 = 0; stage small param vectors in LDS
  if (tid < F_) { xmS[tid] = x_mean[tid]; wdS[tid] = w_dx[tid]; bdS[tid] = b_dx[tid]; }
  for (int e = tid; e < MB_ * F_; e += 512) xl[e] = x_mean[e & (F_ - 1)];
  for (int e = tid; e < MB_ * H_; e += 512) { h0u[e] = 0; h1u[e] = 0; }
  __syncthreads();

  const size_t chan = (size_t)T_ * F_;     // stride between input channels
  for (int t = 0; t < T_; ++t) {
    // ---- elementwise: gamma_x, x_hat, x_last scan; stage bf16 A-matrices --
    for (int e = tid; e < MB_ * F_; e += 512) {
      int bb = bo * MB_ + (e >> 7), f = e & (F_ - 1);
      size_t base = (size_t)bb * 3u * chan + (size_t)t * F_ + f;
      float x  = input[base];
      float mv = input[base + chan];
      float dv = input[base + 2 * chan];
      if (t + 1 < T_) {                       // prefetch next timestep rows
        __builtin_prefetch(input + base + F_, 0, 3);
        __builtin_prefetch(input + base + chan + F_, 0, 3);
        __builtin_prefetch(input + base + 2 * chan + F_, 0, 3);
      }
      float gx = __expf(-fmaxf(dv * wdS[f] + bdS[f], 0.0f));
      float xlast = xl[e];
      float xm = xmS[f];
      float xhat = mv * x + (1.0f - mv) * (gx * xlast + (1.0f - gx) * xm);
      xhu[e] = f2bf(xhat);
      mu[e]  = f2bf(mv);
      du[e]  = f2bf(dv);
      xl[e]  = (mv > 0.0f) ? x : xlast;
    }
    __syncthreads();

    // ---- layer 0: x_in = x_hat (K=128); h0u overwritten with h0_new ------
    grud_layer<128>(xhu, mu, du, h0u, hdu, rhu, Wdh0p, W0p, V0p, U0p,
                    bdh0v, b0r, b0z, b0t, nullptr, wave, lane);
    // ---- layer 1: x_in = h0_new (K=256); writes ys[t] slice --------------
    grud_layer<256>(h0u, mu, du, h1u, hdu, rhu, Wdh1p, W1p, V1p, U1p,
                    bdh1v, b1r, b1z, b1t,
                    out + (size_t)t * B_ * H_ + (size_t)bo * MB_ * H_, wave, lane);
  }
}

// ---------------------------------------------------------------------------
extern "C" void kernel_launch(void* const* d_in, const int* in_sizes, int n_in,
                              void* d_out, int out_size, void* d_ws, size_t ws_size,
                              hipStream_t stream) {
  (void)in_sizes; (void)n_in; (void)out_size; (void)ws_size;
  const float* input  = (const float*)d_in[0];
  const float* x_mean = (const float*)d_in[1];
  const float* w_dx   = (const float*)d_in[2];
  const float* b_dx   = (const float*)d_in[3];
  const float* W_dh0  = (const float*)d_in[4];
  const float* b_dh0  = (const float*)d_in[5];
  const float* W0     = (const float*)d_in[6];
  const float* U0     = (const float*)d_in[7];
  const float* V0     = (const float*)d_in[8];
  const float* b0     = (const float*)d_in[9];
  const float* W_dh1  = (const float*)d_in[10];
  const float* b_dh1  = (const float*)d_in[11];
  const float* W1     = (const float*)d_in[12];
  const float* U1     = (const float*)d_in[13];
  const float* V1     = (const float*)d_in[14];
  const float* b1     = (const float*)d_in[15];
  unsigned* ws = (unsigned*)d_ws;
  float* out = (float*)d_out;

  // pack weights -> bf16 WMMA B-tiles in workspace (cheap, fully parallel)
  pack_bf16_tiles<<<64,  256, 0, stream>>>(W_dh0, 128, 256, ws + OFF_WDH0);
  pack_bf16_tiles<<<192, 256, 0, stream>>>(W0,    128, 768, ws + OFF_W0);
  pack_bf16_tiles<<<192, 256, 0, stream>>>(V0,    128, 768, ws + OFF_V0);
  pack_bf16_tiles<<<384, 256, 0, stream>>>(U0,    256, 768, ws + OFF_U0);
  pack_bf16_tiles<<<64,  256, 0, stream>>>(W_dh1, 128, 256, ws + OFF_WDH1);
  pack_bf16_tiles<<<384, 256, 0, stream>>>(W1,    256, 768, ws + OFF_W1);
  pack_bf16_tiles<<<192, 256, 0, stream>>>(V1,    128, 768, ws + OFF_V1);
  pack_bf16_tiles<<<384, 256, 0, stream>>>(U1,    256, 768, ws + OFF_U1);

  grud_seq<<<2, 512, 0, stream>>>(input, x_mean, w_dx, b_dx, b_dh0, b0,
                                  b_dh1, b1, ws, out);
}
